// TimeGRU_23716809408753
// MI455X (gfx1250) — matrix operand order repaired and checked by reading, not compiled, not run on hardware
//
#include <hip/hip_runtime.h>
#include <hip/hip_bf16.h>

// ---------------- problem constants ----------------
#define B_TOT 4096
#define S_LEN 200
#define I_DIM 128
#define H_DIM 128
#define G3    384            // 3*H
#define M_TILE 16            // batch rows per workgroup
#define NTHREADS 256         // 8 waves (wave32)

// padded LDS strides (bank-conflict avoidance, keep 16B alignment)
#define AK 136               // bf16 K-stride for A buffers (272 B rows)
#define WK 136               // bf16 K-stride for weight buffers
#define XS 388               // f32 stride for xp/hp gate buffers

typedef __bf16 bf16_t;
typedef bf16_t v16bf __attribute__((ext_vector_type(16)));
typedef bf16_t v8bf  __attribute__((ext_vector_type(8)));
typedef float  v8f   __attribute__((ext_vector_type(8)));
typedef float  v4f   __attribute__((ext_vector_type(4)));   // clang vector (NT builtins reject HIP float4)

// ---------------- LDS layout (bytes, all offsets 16B-aligned) ----------------
constexpr int OFF_WIH   = 0;                                  // 384*136*2 = 104448
constexpr int OFF_WHH   = OFF_WIH   + G3 * WK * 2;            // 104448
constexpr int OFF_BIH   = OFF_WHH   + G3 * WK * 2;            // 208896
constexpr int OFF_BHH   = OFF_BIH   + G3 * 4;                 // 210432
constexpr int OFF_XA    = OFF_BHH   + G3 * 4;                 // 211968  (bf16 input tile)
constexpr int OFF_CADJA = OFF_XA    + M_TILE * AK * 2;        // 216320  (bf16 c_adj tile)
constexpr int OFF_CF    = OFF_CADJA + M_TILE * AK * 2;        // 220672  (f32 c state)
constexpr int OFF_CADJF = OFF_CF    + M_TILE * H_DIM * 4;     // 228864  (f32 c_adj)
constexpr int OFF_XP    = OFF_CADJF + M_TILE * H_DIM * 4;     // 237056  (f32 x_proj 16x384 padded)
constexpr int OFF_HP    = OFF_XP    + M_TILE * XS * 4;        // 261888  (f32 h_proj)
constexpr int SMEM_BYTES = OFF_HP   + M_TILE * XS * 4;        // 286720  < 320 KB
static_assert(SMEM_BYTES <= 320 * 1024, "exceeds WGP LDS");

// ---------------- helpers ----------------
__device__ __forceinline__ bf16_t f2bf(float f) {
  unsigned u = __builtin_bit_cast(unsigned, f);
  unsigned r = u + 0x7FFFu + ((u >> 16) & 1u);   // round-to-nearest-even
  return __builtin_bit_cast(bf16_t, (unsigned short)(r >> 16));
}
__device__ __forceinline__ float fast_sigmoid(float x) {
  return 1.0f / (1.0f + __expf(-x));
}
__device__ __forceinline__ float fast_tanh(float x) {
  float e = __expf(2.0f * x);
  return (e - 1.0f) / (e + 1.0f);
}

// A fragment (16x32 bf16, M x K) from row-major [M][AK] LDS buffer.
// Lane L<16: rows M=L, K = k0+0..7 and k0+16..23 ; lanes 16..31: K shifted by 8.
// Each half is 16 contiguous bytes -> ds_load_b128.
__device__ __forceinline__ v16bf load_a_frag(const bf16_t* abuf, int lane, int k0) {
  int mrow = lane & 15;
  int kb   = (lane >> 4) << 3;                   // 0 or 8
  const bf16_t* p = abuf + mrow * AK + k0 + kb;
  v8bf lo = *(const v8bf*)(p);                   // K = k0+kb .. +7
  v8bf hi = *(const v8bf*)(p + 16);              // K = k0+kb+16 .. +23
  v16bf r;
#pragma unroll
  for (int i = 0; i < 8; ++i) { r[i] = lo[i]; r[i + 8] = hi[i]; }
  return r;
}

// B fragment (32x16 bf16, K x N) from [N][WK] row-major weight LDS.
// Lane L<16: col N=L, K = k0+0..7 and k0+16..23 ; lanes 16..31: K shifted by 8.
// Each half is 16 contiguous bytes -> 2x ds_load_b128.
__device__ __forceinline__ v16bf load_b_frag(const bf16_t* w, int lane, int n0, int k0) {
  int n  = lane & 15;
  int ks = (lane >> 4) << 3;                     // 0 or 8
  const bf16_t* p = w + (n0 + n) * WK + k0 + ks;
  v8bf lo = *(const v8bf*)(p);
  v8bf hi = *(const v8bf*)(p + 16);
  v16bf r;
#pragma unroll
  for (int i = 0; i < 8; ++i) { r[i] = lo[i]; r[i + 8] = hi[i]; }
  return r;
}

extern __shared__ unsigned char smem_raw[];

// ---------------- fused persistent TimeGRU kernel ----------------
__global__ __launch_bounds__(NTHREADS, 1)
void timegru_fused_kernel(const float* __restrict__ inputs,
                          const float* __restrict__ time_diff,
                          const int*   __restrict__ mask,
                          const float* __restrict__ W_ih,
                          const float* __restrict__ W_hh,
                          const float* __restrict__ b_ih,
                          const float* __restrict__ b_hh,
                          float* __restrict__ outputs,
                          float* __restrict__ cfin0,
                          float* __restrict__ cfin1) {
  const int tid  = threadIdx.x;
  const int lane = tid & 31;
  const int wv   = tid >> 5;                     // wave id 0..7
  const int b0   = blockIdx.x * M_TILE;

  bf16_t* sWih   = (bf16_t*)(smem_raw + OFF_WIH);
  bf16_t* sWhh   = (bf16_t*)(smem_raw + OFF_WHH);
  float*  sBih   = (float*)(smem_raw + OFF_BIH);
  float*  sBhh   = (float*)(smem_raw + OFF_BHH);
  bf16_t* sXA    = (bf16_t*)(smem_raw + OFF_XA);
  bf16_t* sCadjA = (bf16_t*)(smem_raw + OFF_CADJA);
  float*  sC     = (float*)(smem_raw + OFF_CF);
  float*  sCadjF = (float*)(smem_raw + OFF_CADJF);
  float*  sXP    = (float*)(smem_raw + OFF_XP);
  float*  sHP    = (float*)(smem_raw + OFF_HP);

  // ---- one-time: weights -> bf16 LDS (resident for all 200 steps) ----
  for (int e = tid; e < G3 * I_DIM; e += NTHREADS) {
    int rw = e >> 7, cl = e & 127;
    sWih[rw * WK + cl] = f2bf(W_ih[e]);
    sWhh[rw * WK + cl] = f2bf(W_hh[e]);
  }
  for (int e = tid; e < G3; e += NTHREADS) { sBih[e] = b_ih[e]; sBhh[e] = b_hh[e]; }
  for (int e = tid; e < M_TILE * H_DIM; e += NTHREADS) sC[e] = 0.0f;

  const int m  = tid >> 4;                       // 0..15   batch row within tile
  const int k0 = (tid & 15) << 3;                // 0..120  8 consecutive K/H per thread

  const size_t rowBS = (size_t)(b0 + m) * S_LEN; // row base into [B,S] tensors
  const float* xbase = inputs + rowBS * (size_t)I_DIM + k0;

  // ---- software prefetch registers (hide HBM latency behind WMMA phases) ----
  // Streaming, read-once data: use non-temporal loads to keep L2 clean.
  v4f px0 = __builtin_nontemporal_load((const v4f*)(xbase));
  v4f px1 = __builtin_nontemporal_load((const v4f*)(xbase + 4));
  float ptd = __builtin_nontemporal_load(time_diff + rowBS);
  int   pmk = __builtin_nontemporal_load(mask + rowBS);

  for (int t = 0; t < S_LEN; ++t) {
    __syncthreads();   // previous step's phase C done before rewriting A buffers

    // ---------- Phase A: stage inputs (bf16) + decayed state c_adj ----------
    const int mk_cur = pmk;
    {
      bf16_t* xa = sXA + m * AK + k0;
#pragma unroll
      for (int j = 0; j < 4; ++j) { xa[j] = f2bf(px0[j]); xa[4 + j] = f2bf(px1[j]); }

      float tdw = 1.0f / __logf(2.71828182845904523536f + ptd);   // 1/log(e+dt)
#pragma unroll
      for (int j = 0; j < 8; ++j) {
        float ca = sC[m * H_DIM + k0 + j] * tdw;
        sCadjF[m * H_DIM + k0 + j] = ca;
        sCadjA[m * AK + k0 + j]    = f2bf(ca);
      }
    }

    // ---------- Prefetch step t+1 (overlaps with phases B + C) ----------
    if (t + 1 < S_LEN) {
      const float* xn = xbase + (size_t)(t + 1) * I_DIM;
      px0 = __builtin_nontemporal_load((const v4f*)(xn));
      px1 = __builtin_nontemporal_load((const v4f*)(xn + 4));
      ptd = __builtin_nontemporal_load(time_diff + rowBS + t + 1);
      pmk = __builtin_nontemporal_load(mask + rowBS + t + 1);
    }
    __syncthreads();

    // ---------- Phase B: dual WMMA GEMMs  xp = x*Wih^T ; hp = c_adj*Whh^T ----------
    {
      v16bf ax[4], ah[4];
#pragma unroll
      for (int kk = 0; kk < 4; ++kk) {
        ax[kk] = load_a_frag(sXA,    lane, kk * 32);
        ah[kk] = load_a_frag(sCadjA, lane, kk * 32);
      }
#pragma unroll
      for (int i = 0; i < 3; ++i) {              // 24 N-tiles / 8 waves = 3 each
        int n0 = (wv * 3 + i) * 16;
        v8f accx = {};
        v8f acch = {};
#pragma unroll
        for (int kk = 0; kk < 4; ++kk) {
          v16bf bi = load_b_frag(sWih, lane, n0, kk * 32);
          accx = __builtin_amdgcn_wmma_f32_16x16x32_bf16(false, ax[kk], false, bi,
                                                         (short)0, accx, false, false);
          v16bf bh = load_b_frag(sWhh, lane, n0, kk * 32);
          acch = __builtin_amdgcn_wmma_f32_16x16x32_bf16(false, ah[kk], false, bh,
                                                         (short)0, acch, false, false);
        }
        int col = n0 + (lane & 15);
        int r0  = (lane >> 4) * 8;               // C/D layout: VGPR j -> M=j (+8 for hi lanes)
#pragma unroll
        for (int j = 0; j < 8; ++j) {
          sXP[(r0 + j) * XS + col] = accx[j];
          sHP[(r0 + j) * XS + col] = acch[j];
        }
      }
    }
    __syncthreads();

    // ---------- Phase C: gates, state update, masked hold, output store ----------
    {
      float cn[8];
#pragma unroll
      for (int j = 0; j < 8; ++j) {
        int h = k0 + j;
        float xr = sXP[m * XS + h]       + sBih[h];
        float hr = sHP[m * XS + h]       + sBhh[h];
        float xz = sXP[m * XS + 128 + h] + sBih[128 + h];
        float hz = sHP[m * XS + 128 + h] + sBhh[128 + h];
        float xn = sXP[m * XS + 256 + h] + sBih[256 + h];
        float hn = sHP[m * XS + 256 + h] + sBhh[256 + h];
        float r = fast_sigmoid(xr + hr);
        float z = fast_sigmoid(xz + hz);
        float n = fast_tanh(xn + r * hn);
        float ca   = sCadjF[m * H_DIM + h];
        float cold = sC[m * H_DIM + h];
        float v = (1.0f - z) * n + z * ca;
        cn[j] = mk_cur ? cold : v;               // masked rows keep pre-decay state
      }
#pragma unroll
      for (int j = 0; j < 8; ++j) sC[m * H_DIM + k0 + j] = cn[j];

      // streaming write-once output: non-temporal stores keep L2 clean
      float* orow = outputs + (rowBS + t) * (size_t)H_DIM + k0;
      v4f o0 = {cn[0], cn[1], cn[2], cn[3]};
      v4f o1 = {cn[4], cn[5], cn[6], cn[7]};
      __builtin_nontemporal_store(o0, (v4f*)(orow));
      __builtin_nontemporal_store(o1, (v4f*)(orow + 4));
    }
  }
  __syncthreads();

  // ---- final state, duplicated (reference returns (c_final, c_final)) ----
  for (int e = tid; e < M_TILE * H_DIM; e += NTHREADS) {
    int rw = e >> 7, h = e & 127;
    float v = sC[e];
    cfin0[(size_t)(b0 + rw) * H_DIM + h] = v;
    cfin1[(size_t)(b0 + rw) * H_DIM + h] = v;
  }
}

// ---------------- host launch ----------------
extern "C" void kernel_launch(void* const* d_in, const int* in_sizes, int n_in,
                              void* d_out, int out_size, void* d_ws, size_t ws_size,
                              hipStream_t stream) {
  (void)in_sizes; (void)n_in; (void)out_size; (void)d_ws; (void)ws_size;
  const float* inputs    = (const float*)d_in[0];   // [B,S,I]
  const float* time_diff = (const float*)d_in[1];   // [B,S]
  const int*   mask      = (const int*)  d_in[2];   // [B,S]
  const float* W_ih      = (const float*)d_in[3];   // [3H,I]
  const float* W_hh      = (const float*)d_in[4];   // [3H,H]
  const float* b_ih      = (const float*)d_in[5];   // [3H]
  const float* b_hh      = (const float*)d_in[6];   // [3H]

  float* out    = (float*)d_out;
  float* outs   = out;                                        // [B,S,H]
  float* cfin0  = out + (size_t)B_TOT * S_LEN * H_DIM;        // [B,H]
  float* cfin1  = cfin0 + (size_t)B_TOT * H_DIM;              // [B,H]

  // allow >64KB dynamic LDS (280 KB of the WGP's 320 KB)
  static_cast<void>(hipFuncSetAttribute((const void*)timegru_fused_kernel,
                                        hipFuncAttributeMaxDynamicSharedMemorySize,
                                        SMEM_BYTES));

  dim3 grid(B_TOT / M_TILE);   // 256 workgroups
  dim3 block(NTHREADS);        // 8 waves of 32
  timegru_fused_kernel<<<grid, block, SMEM_BYTES, stream>>>(
      inputs, time_diff, mask, W_ih, W_hh, b_ih, b_hh, outs, cfin0, cfin1);
}
